// Decoder_75823352644142
// MI455X (gfx1250) — matrix operand order repaired and checked by reading, not compiled, not run on hardware
//
#include <hip/hip_runtime.h>

// ---------------------------------------------------------------------------
// Decoder step for MI455X (gfx1250, wave32, WMMA).
// Heavy op: scores = tanh(q + enc @ uW.T) @ v  -> bf16 WMMA GEMM with
// double-buffered LDS staging (global->LDS latency hidden behind WMMAs).
// Everything else is bandwidth-bound GEMV / elementwise.
// ---------------------------------------------------------------------------

#define HH 1024
#define LL 4096
#define VV 32000

typedef __attribute__((ext_vector_type(16))) __bf16  v16bf;
typedef __attribute__((ext_vector_type(8)))  float   v8f;
typedef __attribute__((ext_vector_type(4)))  float   float4v;
typedef __attribute__((ext_vector_type(4)))  unsigned int uint4v;
typedef __attribute__((ext_vector_type(2)))  unsigned int uint2v;

union FragBF { uint4v q[2]; v16bf v; };   // 32B: 16 bf16 fragment

__device__ __forceinline__ unsigned short f2bf(float x) {
  union { float f; unsigned int u; } c; c.f = x;
  unsigned int u = c.u;
  u += 0x7FFFu + ((u >> 16) & 1u);        // round-to-nearest-even
  return (unsigned short)(u >> 16);
}

// ---------------------------------------------------------------------------
// Kernel 1: q[j] = wW[j,:] . hidden + wb[j] + ub[j]     (grid 4 x 256)
// ---------------------------------------------------------------------------
__global__ __launch_bounds__(256) void q_kernel(
    const float* __restrict__ wW, const float* __restrict__ wb,
    const float* __restrict__ ub, const float* __restrict__ hidden,
    float* __restrict__ q) {
  __shared__ float hsh[HH];
  for (int i = threadIdx.x; i < HH; i += 256) hsh[i] = hidden[i];
  __syncthreads();
  int j = blockIdx.x * 256 + threadIdx.x;
  const float4v* row = (const float4v*)(wW + (size_t)j * HH);
  const float4v* hv  = (const float4v*)hsh;
  float acc = 0.f;
  for (int k = 0; k < HH / 4; ++k) {
    float4v w = row[k], x = hv[k];
    acc += w.x * x.x + w.y * x.y + w.z * x.z + w.w * x.w;
  }
  q[j] = acc + wb[j] + ub[j];
}

// ---------------------------------------------------------------------------
// Kernel 2: attention scores via bf16 WMMA.
//   scores[l] = sum_j vW[j] * tanh(q[j] + enc[l,:] . uW[j,:])
// Grid: 128 WGs x 256 threads (8 wave32s). WG owns 32 enc rows (M).
// Waves: msub = wave&1 (M subtile), nq = wave>>1 (N quarter).
// LDS: A = 32 rows x 1024 K bf16, row stride 1032 (4-bank row shift,
//        conflict-free 16B fragment reads).
//      B = double-buffered: 2 x (4 n-tile slots x 16 rows x 256-K chunk),
//        row stride 272 (8-bank row shift, optimal 32B fragment reads).
// Pipeline: stage chunk i+1 while WMMAs consume chunk i; one barrier/iter.
// ---------------------------------------------------------------------------
#define ASTR 1032
#define BKC  256
#define BSTR 272
#define BBUF (4 * 16 * BSTR)

__global__ __launch_bounds__(256) void attn_scores_kernel(
    const float* __restrict__ enc, const float* __restrict__ uW,
    const float* __restrict__ q,   const float* __restrict__ vW,
    float* __restrict__ scores) {
  __shared__ __align__(16) unsigned short lA[32 * ASTR];
  __shared__ __align__(16) unsigned short lB[2 * BBUF];
  __shared__ float psum[3 * 32];

  const int tid  = threadIdx.x;
  const int wave = tid >> 5;
  const int lane = tid & 31;
  const int msub = wave & 1;        // M subtile (0..1)
  const int nq   = wave >> 1;       // N quarter (0..3)
  const int hi   = lane >> 4;
  const int ln   = lane & 15;
  const int mb   = blockIdx.x * 32;

  // Cooperative B-chunk stage: 4 tiles x 16 rows x 256 K of uW -> bf16 LDS.
  auto stageB = [&](int it, int kc, int buf) {
    unsigned short* dst = &lB[buf * BBUF];
#pragma unroll
    for (int i0 = 0; i0 < 16; ++i0) {
      int i  = tid + i0 * 256;
      int t  = i >> 10;                 // tile slot 0..3
      int rr = (i >> 6) & 15;
      int c4 = i & 63;
      int j  = ((t * 16 + it) << 4) + rr;     // uW row = output column n
      float4v f = ((const float4v*)(uW + (size_t)j * HH + kc * BKC))[c4];
      uint2v pk;
      pk.x = (unsigned int)f2bf(f.x) | ((unsigned int)f2bf(f.y) << 16);
      pk.y = (unsigned int)f2bf(f.z) | ((unsigned int)f2bf(f.w) << 16);
      *(uint2v*)&dst[(t * 16 + rr) * BSTR + c4 * 4] = pk;
    }
  };

  // Stage A: 32 rows x 1024 f32 -> bf16 LDS (once per WG).
  for (int i = tid; i < 32 * (HH / 4); i += 256) {
    int r  = i >> 8;
    int c4 = i & 255;
    float4v f = ((const float4v*)(enc + (size_t)(mb + r) * HH))[c4];
    uint2v pk;
    pk.x = (unsigned int)f2bf(f.x) | ((unsigned int)f2bf(f.y) << 16);
    pk.y = (unsigned int)f2bf(f.z) | ((unsigned int)f2bf(f.w) << 16);
    *(uint2v*)&lA[r * ASTR + c4 * 4] = pk;
  }
  stageB(0, 0, 0);                  // prologue: first B chunk
  __syncthreads();

  const unsigned short* Abase = &lA[(msub * 16 + ln) * ASTR];

  float sacc[8];
#pragma unroll
  for (int r = 0; r < 8; ++r) sacc[r] = 0.f;

  int pbuf = 0;
  for (int it = 0; it < 16; ++it) {       // this wave's n-tile: nq*16 + it
    v8f c = {};
    for (int kc = 0; kc < 4; ++kc) {
      // Prefetch next chunk into the other buffer (overlaps the WMMAs).
      if (!(it == 15 && kc == 3)) {
        int nit = (kc == 3) ? it + 1 : it;
        int nkc = (kc + 1) & 3;
        stageB(nit, nkc, pbuf ^ 1);
      }

      const unsigned short* Bbase = &lB[pbuf * BBUF + (nq * 16 + ln) * BSTR];
#pragma unroll
      for (int ks = 0; ks < 8; ++ks) {
        int kg = kc * BKC + ks * 32;      // global K (A)
        int kl = ks * 32;                 // local K (B chunk)
        FragBF fa, fb;
        // A 16x32 bf16 layout: lane<16 holds K 0..7 & 16..23 of row ln;
        // lane>=16 holds K 8..15 & 24..31.
        fa.q[0] = *(const uint4v*)&Abase[kg + 8 * hi];
        fa.q[1] = *(const uint4v*)&Abase[kg + 16 + 8 * hi];
        // B 32x16: lane n=ln; lane<16 holds K 0..7 & 16..23, lane>=16 K 8..15 & 24..31.
        fb.q[0] = *(const uint4v*)&Bbase[kl + 8 * hi];
        fb.q[1] = *(const uint4v*)&Bbase[kl + 16 + 8 * hi];
        c = __builtin_amdgcn_wmma_f32_16x16x32_bf16(
                false, fa.v, false, fb.v, (short)0, c, false, false);
      }
      // Single barrier per iteration: next-chunk stores complete, and the
      // buffer just consumed is only overwritten after the *next* barrier.
      __syncthreads();
      pbuf ^= 1;
    }
    // Fused epilogue: C tile lane holds col n=ln, rows r + 8*hi.
    int n = ((nq * 16 + it) << 4) + ln;
    float vn = vW[n];
    float qn = q[n];
#pragma unroll
    for (int r = 0; r < 8; ++r)
      sacc[r] += vn * tanhf(qn + c[r]);
  }

  // Reduce across the 16 lanes (different n) within each half-wave.
  float red[8];
#pragma unroll
  for (int r = 0; r < 8; ++r) {
    float s = sacc[r];
    s += __shfl_xor(s, 8, 16);
    s += __shfl_xor(s, 4, 16);
    s += __shfl_xor(s, 2, 16);
    s += __shfl_xor(s, 1, 16);
    red[r] = s;
  }
  // Combine the 4 N-quarter waves through LDS.
  if (nq != 0 && ln == 0) {
#pragma unroll
    for (int r = 0; r < 8; ++r)
      psum[(nq - 1) * 32 + msub * 16 + r + 8 * hi] = red[r];
  }
  __syncthreads();
  if (nq == 0 && ln == 0) {
#pragma unroll
    for (int r = 0; r < 8; ++r) {
      int ri = msub * 16 + r + 8 * hi;
      scores[mb + ri] = red[r] + psum[ri] + psum[32 + ri] + psum[64 + ri];
    }
  }
}

// ---------------------------------------------------------------------------
// Kernel 3: softmax over L=4096 (1 WG x 1024 threads, 4 elems/thread)
// ---------------------------------------------------------------------------
__global__ __launch_bounds__(1024) void softmax_kernel(
    const float* __restrict__ scores, float* __restrict__ attn) {
  __shared__ float red[32];
  __shared__ float bc[2];
  int tid  = threadIdx.x;
  int lane = tid & 31, wid = tid >> 5;
  float v0 = scores[tid],        v1 = scores[tid + 1024];
  float v2 = scores[tid + 2048], v3 = scores[tid + 3072];
  float m = fmaxf(fmaxf(v0, v1), fmaxf(v2, v3));
  for (int s = 16; s > 0; s >>= 1) m = fmaxf(m, __shfl_xor(m, s, 32));
  if (lane == 0) red[wid] = m;
  __syncthreads();
  if (wid == 0) {
    float t = red[lane];
    for (int s = 16; s > 0; s >>= 1) t = fmaxf(t, __shfl_xor(t, s, 32));
    if (lane == 0) bc[0] = t;
  }
  __syncthreads();
  m = bc[0];
  float e0 = expf(v0 - m), e1 = expf(v1 - m);
  float e2 = expf(v2 - m), e3 = expf(v3 - m);
  float s = e0 + e1 + e2 + e3;
  for (int sh = 16; sh > 0; sh >>= 1) s += __shfl_xor(s, sh, 32);
  __syncthreads();
  if (lane == 0) red[wid] = s;
  __syncthreads();
  if (wid == 0) {
    float t = red[lane];
    for (int sh = 16; sh > 0; sh >>= 1) t += __shfl_xor(t, sh, 32);
    if (lane == 0) bc[1] = t;
  }
  __syncthreads();
  float inv = 1.0f / bc[1];
  attn[tid]        = e0 * inv;
  attn[tid + 1024] = e1 * inv;
  attn[tid + 2048] = e2 * inv;
  attn[tid + 3072] = e3 * inv;
}

// ---------------------------------------------------------------------------
// Kernel 4: context[h] = sum_l attn[l] * enc[l,h]   (grid 4 x 256, coalesced)
// ---------------------------------------------------------------------------
__global__ __launch_bounds__(256) void context_kernel(
    const float* __restrict__ enc, const float* __restrict__ attn,
    float* __restrict__ ctx) {
  int h = blockIdx.x * 256 + threadIdx.x;
  float acc = 0.f;
  for (int l = 0; l < LL; l += 4) {
    float a0 = attn[l], a1 = attn[l + 1], a2 = attn[l + 2], a3 = attn[l + 3];
    acc += a0 * enc[(size_t)l       * HH + h];
    acc += a1 * enc[(size_t)(l + 1) * HH + h];
    acc += a2 * enc[(size_t)(l + 2) * HH + h];
    acc += a3 * enc[(size_t)(l + 3) * HH + h];
  }
  ctx[h] = acc;
}

// ---------------------------------------------------------------------------
// Kernel 5: x = relu([ctx; emb[token]] @ cW.T + cb)   (grid 4 x 256)
// ---------------------------------------------------------------------------
__global__ __launch_bounds__(256) void combine_kernel(
    const float* __restrict__ cW, const float* __restrict__ cb,
    const float* __restrict__ ctx, const float* __restrict__ emb,
    const int* __restrict__ token, float* __restrict__ x) {
  __shared__ float xin[2 * HH];
  int t0 = token[0];
  for (int i = threadIdx.x; i < HH; i += 256) {
    xin[i]      = ctx[i];
    xin[HH + i] = emb[(size_t)t0 * HH + i];
  }
  __syncthreads();
  int j = blockIdx.x * 256 + threadIdx.x;
  const float4v* row = (const float4v*)(cW + (size_t)j * 2 * HH);
  const float4v* xv  = (const float4v*)xin;
  float acc = 0.f;
  for (int k = 0; k < (2 * HH) / 4; ++k) {
    float4v w = row[k], v = xv[k];
    acc += w.x * v.x + w.y * v.y + w.z * v.z + w.w * v.w;
  }
  x[j] = fmaxf(acc + cb[j], 0.f);
}

// ---------------------------------------------------------------------------
// Kernel 6: GRU gate GEMVs: gi = x @ w_ih.T + b_ih; gh = h @ w_hh.T + b_hh
// (grid 12 x 256, one row of each matrix per thread)
// ---------------------------------------------------------------------------
__global__ __launch_bounds__(256) void gru_gates_kernel(
    const float* __restrict__ w_ih, const float* __restrict__ b_ih,
    const float* __restrict__ w_hh, const float* __restrict__ b_hh,
    const float* __restrict__ x, const float* __restrict__ hidden,
    float* __restrict__ gi, float* __restrict__ gh) {
  __shared__ float xs[HH], hs[HH];
  for (int i = threadIdx.x; i < HH; i += 256) { xs[i] = x[i]; hs[i] = hidden[i]; }
  __syncthreads();
  int i = blockIdx.x * 256 + threadIdx.x;    // 0..3071
  const float4v* wi = (const float4v*)(w_ih + (size_t)i * HH);
  const float4v* wh = (const float4v*)(w_hh + (size_t)i * HH);
  const float4v* xv = (const float4v*)xs;
  const float4v* hv = (const float4v*)hs;
  float ai = 0.f, ah = 0.f;
  for (int k = 0; k < HH / 4; ++k) {
    float4v a = wi[k], b = wh[k], u = xv[k], v = hv[k];
    ai += a.x * u.x + a.y * u.y + a.z * u.z + a.w * u.w;
    ah += b.x * v.x + b.y * v.y + b.z * v.z + b.w * v.w;
  }
  gi[i] = ai + b_ih[i];
  gh[i] = ah + b_hh[i];
}

// ---------------------------------------------------------------------------
// Kernel 7: GRU elementwise finalize + residual vector   (grid 4 x 256)
// ---------------------------------------------------------------------------
__global__ __launch_bounds__(256) void gru_final_kernel(
    const float* __restrict__ gi, const float* __restrict__ gh,
    const float* __restrict__ hidden, const float* __restrict__ ctx,
    float* __restrict__ new_hidden, float* __restrict__ ovec) {
  int h = blockIdx.x * 256 + threadIdx.x;
  float r = 1.f / (1.f + expf(-(gi[h] + gh[h])));
  float z = 1.f / (1.f + expf(-(gi[HH + h] + gh[HH + h])));
  float n = tanhf(gi[2 * HH + h] + r * gh[2 * HH + h]);
  float hp = hidden[h];
  float hn = (1.f - z) * n + z * hp;
  new_hidden[h] = hn;
  ovec[h] = ctx[h] + hn;
}

// ---------------------------------------------------------------------------
// Kernel 8: logits = ovec @ outW.T + outb  (128 MB -> pure bandwidth)
// grid 250 x 256: 2 lanes per output row (split-K 512), shuffle combine.
// ---------------------------------------------------------------------------
__global__ __launch_bounds__(256) void logits_kernel(
    const float* __restrict__ outW, const float* __restrict__ outb,
    const float* __restrict__ ovec, float* __restrict__ logits) {
  __shared__ float vs[HH];
  for (int i = threadIdx.x; i < HH; i += 256) vs[i] = ovec[i];
  __syncthreads();
  int gid  = blockIdx.x * 256 + threadIdx.x;
  int v    = gid >> 1;
  int half = gid & 1;
  const float4v* row = (const float4v*)(outW + (size_t)v * HH + half * (HH / 2));
  const float4v* xv  = ((const float4v*)vs) + half * (HH / 8);
  float acc = 0.f;
  for (int k = 0; k < HH / 8; ++k) {
    float4v w = row[k], x = xv[k];
    acc += w.x * x.x + w.y * x.y + w.z * x.z + w.w * x.w;
  }
  acc += __shfl_xor(acc, 1, 32);
  if (half == 0) logits[v] = acc + outb[v];
}

// ---------------------------------------------------------------------------
extern "C" void kernel_launch(void* const* d_in, const int* in_sizes, int n_in,
                              void* d_out, int out_size, void* d_ws, size_t ws_size,
                              hipStream_t stream) {
  const int*   token  = (const int*)  d_in[0];
  const float* hidden = (const float*)d_in[1];
  const float* enc    = (const float*)d_in[2];
  const float* emb    = (const float*)d_in[3];
  const float* wW     = (const float*)d_in[4];
  const float* wb     = (const float*)d_in[5];
  const float* uW     = (const float*)d_in[6];
  const float* ub     = (const float*)d_in[7];
  const float* vW     = (const float*)d_in[8];
  const float* cW     = (const float*)d_in[9];
  const float* cb     = (const float*)d_in[10];
  const float* w_ih   = (const float*)d_in[11];
  const float* b_ih   = (const float*)d_in[12];
  const float* w_hh   = (const float*)d_in[13];
  const float* b_hh   = (const float*)d_in[14];
  const float* outW   = (const float*)d_in[15];
  const float* outb   = (const float*)d_in[16];

  float* logits     = (float*)d_out;        // [V]
  float* new_hidden = logits + VV;          // [H]
  float* attn       = new_hidden + HH;      // [L]

  float* ws     = (float*)d_ws;
  float* q      = ws;             // 1024
  float* scores = ws + 1024;      // 4096
  float* ctx    = ws + 5120;      // 1024
  float* x      = ws + 6144;      // 1024
  float* gi     = ws + 7168;      // 3072
  float* gh     = ws + 10240;     // 3072
  float* ovec   = ws + 13312;     // 1024

  q_kernel          <<<4,   256, 0, stream>>>(wW, wb, ub, hidden, q);
  attn_scores_kernel<<<128, 256, 0, stream>>>(enc, uW, q, vW, scores);
  softmax_kernel    <<<1,  1024, 0, stream>>>(scores, attn);
  context_kernel    <<<4,   256, 0, stream>>>(enc, attn, ctx);
  combine_kernel    <<<4,   256, 0, stream>>>(cW, cb, ctx, emb, token, x);
  gru_gates_kernel  <<<12,  256, 0, stream>>>(w_ih, b_ih, w_hh, b_hh, x, hidden, gi, gh);
  gru_final_kernel  <<<4,   256, 0, stream>>>(gi, gh, hidden, ctx, new_hidden, ovec);
  logits_kernel     <<<250, 256, 0, stream>>>(outW, outb, ovec, logits);
}
